// FluxTransformerBlock_16716012716264
// MI455X (gfx1250) — matrix operand order repaired and checked by reading, not compiled, not run on hardware
//
#include <hip/hip_runtime.h>
#include <math.h>

#define HEADS 24
#define HD 128
#define DMODEL 3072
#define S_IMG 2048
#define S_TXT 256
#define S_TOT 2304
#define EPSV 1e-6f

typedef __attribute__((ext_vector_type(16))) __bf16 v16bf;
typedef __attribute__((ext_vector_type(8)))  float v8f;
typedef __attribute__((ext_vector_type(4)))  unsigned int v4u;
typedef __attribute__((ext_vector_type(8)))  int v8i_;
typedef __attribute__((ext_vector_type(4)))  int v4i_;

union U4 { uint4 u; __bf16 h[8]; };

__device__ __forceinline__ float gelu_tanh(float x){
  float x3 = x*x*x;
  return 0.5f*x*(1.f + tanhf(0.7978845608028654f*(x + 0.044715f*x3)));
}

// raw 16B global->LDS async copy (GVS mode: SGPR base + per-lane 32-bit offset)
__device__ __forceinline__ void async_copy_b128(unsigned lds_off, const void* gbase,
                                                unsigned byte_off){
  asm volatile("global_load_async_to_lds_b128 %0, %1, %2 offset:0"
               :: "v"(lds_off), "v"(byte_off),
                  "s"((unsigned long long)(uintptr_t)gbase)
               : "memory");
}
__device__ __forceinline__ void wait_async0(){
  asm volatile("s_wait_asynccnt 0x0" ::: "memory");
}

// TDM: DMA a 64-row x 128-col bf16 tile (row stride 128 elems) into LDS with a
// 16B pad every 256B so the landed pitch is 136 elements (= KPITCH).
// This toolchain's builtin is the 6-arg flavor:
//   (uint32x4 g0, int32x8 g1, int32x4 g2, int32x4 g3, int32x8, i32 cpol)
__device__ __forceinline__ void tdm_load_tile_64x128_bf16(const void* gaddr,
                                                          unsigned lds_off){
  unsigned long long ga = (unsigned long long)(uintptr_t)gaddr;
  v4u g0;
  g0[0] = 1u;                                            // count=1 user D#
  g0[1] = lds_off;                                       // lds_addr
  g0[2] = (unsigned)(ga & 0xffffffffu);                  // global_addr[31:0]
  g0[3] = (unsigned)((ga >> 32) & 0x01ffffffu) | (2u << 30); // addr[56:32]|type=2
  v8i_ g1;
  g1[0] = (int)((1u << 16)          // data_size = 2 bytes
              | (1u << 20)          // pad_enable
              | (5u << 22)          // pad_interval: 64 DWORDs (256B)
              | (3u << 25));        // pad_amount: 4 DWORDs (16B)
  g1[1] = (int)(128u << 16);        // tensor_dim0 = 128 (bits 79:48)
  g1[2] = (int)(64u  << 16);        // tensor_dim1 = 64  (bits 111:80)
  g1[3] = (int)(128u << 16);        // tile_dim0 = 128   (bits 127:112)
  g1[4] = 64;                       // tile_dim1 = 64, tile_dim2 = 0
  g1[5] = 128;                      // tensor_dim0_stride = 128
  g1[6] = 0; g1[7] = 0;             // tensor_dim1_stride = 0 (2D tile)
  v4i_ z4 = {0, 0, 0, 0};
  v8i_ z8 = {0, 0, 0, 0, 0, 0, 0, 0};
  __builtin_amdgcn_tensor_load_to_lds(g0, g1, z4, z4, z8, 0);
}

// ---------------------------------------------------------------------------
__global__ void __launch_bounds__(256)
silu_kernel(const float* __restrict__ in, float* __restrict__ out, int n){
  int i = blockIdx.x*256 + threadIdx.x;
  if (i < n){ float v = in[i]; out[i] = v / (1.f + __expf(-v)); }
}

// ---------------------------------------------------------------------------
__global__ void __launch_bounds__(256)
ada_gemv(const float* __restrict__ z, const float* __restrict__ W,
         const float* __restrict__ b, float* __restrict__ out, int K, int N){
  __shared__ float zs[DMODEL];
  for (int i = threadIdx.x; i < K; i += 256) zs[i] = z[i];
  __syncthreads();
  int c = blockIdx.x*256 + threadIdx.x;
  float s = 0.f;
  for (int k = 0; k < K; ++k) s += zs[k] * W[(size_t)k*N + c];
  out[c] = s + b[c];
}

// ---------------------------------------------------------------------------
__global__ void __launch_bounds__(256)
ln_mod(const float* __restrict__ in, const float* __restrict__ scale,
       const float* __restrict__ shift, float* __restrict__ out){
  const int row = blockIdx.x;
  const float* r = in + (size_t)row*DMODEL;
  float s = 0.f, s2 = 0.f;
  for (int i = threadIdx.x; i < DMODEL; i += 256){ float v = r[i]; s += v; s2 += v*v; }
  #pragma unroll
  for (int m = 16; m >= 1; m >>= 1){ s += __shfl_xor(s, m, 32); s2 += __shfl_xor(s2, m, 32); }
  __shared__ float red1[8], red2[8];
  int lane = threadIdx.x & 31, wv = threadIdx.x >> 5;
  if (lane == 0){ red1[wv] = s; red2[wv] = s2; }
  __syncthreads();
  if (wv == 0){
    s  = (lane < 8) ? red1[lane] : 0.f;
    s2 = (lane < 8) ? red2[lane] : 0.f;
    #pragma unroll
    for (int m = 4; m >= 1; m >>= 1){ s += __shfl_xor(s, m, 32); s2 += __shfl_xor(s2, m, 32); }
    if (lane == 0){ red1[0] = s; red2[0] = s2; }
  }
  __syncthreads();
  float mean = red1[0] * (1.f/DMODEL);
  float var  = red2[0] * (1.f/DMODEL) - mean*mean;
  float rstd = rsqrtf(var + EPSV);
  float* o = out + (size_t)row*DMODEL;
  for (int i = threadIdx.x; i < DMODEL; i += 256){
    float v = (r[i] - mean) * rstd;
    o[i] = v * (1.f + scale[i]) + shift[i];
  }
}

// ---------------------------------------------------------------------------
__global__ void __launch_bounds__(256)
res_gate(const float* __restrict__ base, const float* __restrict__ val,
         const float* __restrict__ gate, float* __restrict__ out,
         long long total, int cols){
  long long i = (long long)blockIdx.x*256 + threadIdx.x;
  if (i >= total) return;
  int c = (int)(i % cols);
  out[i] = base[i] + gate[c] * val[i];
}

// ---------------------------------------------------------------------------
// Tiled bf16 WMMA GEMM with register double-buffered staging
#define BM 128
#define BN 128
#define BK 32
#define APITCH 40
#define BPITCH 40

__device__ __forceinline__ void stage_load(const float* __restrict__ Ag,
                                           const float* __restrict__ Bg,
                                           int K, int N, int blockM, int blockN,
                                           int k0, int tid, float4* fa, float* fbv){
  #pragma unroll
  for (int it = 0; it < 4; ++it){
    int idx = (tid + it*256) * 4;
    int r = idx >> 5, c = idx & 31;
    fa[it] = *(const float4*)(Ag + (size_t)(blockM + r)*K + k0 + c);
  }
  // B: 4 consecutive k-rows x 4 n-cols per thread (enables packed b64 stores)
  int rB = (tid >> 5) * 4, cB = (tid & 31) * 4;
  #pragma unroll
  for (int j = 0; j < 4; ++j){
    float4 f = *(const float4*)(Bg + (size_t)(k0 + rB + j)*N + blockN + cB);
    fbv[j*4+0] = f.x; fbv[j*4+1] = f.y; fbv[j*4+2] = f.z; fbv[j*4+3] = f.w;
  }
}

__global__ void __launch_bounds__(256)
gemm_bf16(const float* __restrict__ Ag, const float* __restrict__ Bg,
          const float* __restrict__ bias, float* __restrict__ Dg,
          int M, int N, int K, int do_gelu){
  __shared__ __bf16 As[BM][APITCH];   // row-major M x K-chunk
  __shared__ __bf16 Bt[BN][BPITCH];   // transposed: N x K-chunk
  const int tid  = threadIdx.x;
  const int lane = tid & 31, wave = tid >> 5;
  const int l16  = lane & 15, hi = lane >> 4;
  const int waveM = wave >> 1;        // 0..3 -> 32 rows each
  const int waveN = wave & 1;         // 0..1 -> 64 cols each
  const int blockM = blockIdx.y * BM, blockN = blockIdx.x * BN;

  v8f acc[2][4];
  #pragma unroll
  for (int i = 0; i < 2; ++i)
    #pragma unroll
    for (int j = 0; j < 4; ++j)
      #pragma unroll
      for (int e = 0; e < 8; ++e) acc[i][j][e] = 0.f;

  float4 fa[4], fa2[4];
  float fbv[16], fbv2[16];
  stage_load(Ag, Bg, K, N, blockM, blockN, 0, tid, fa, fbv);

  const int rB = (tid >> 5)*4, cB = (tid & 31)*4;

  for (int k0 = 0; k0 < K; k0 += BK){
    // current staged registers -> bf16 LDS (A row-major, B transposed packed b64)
    #pragma unroll
    for (int it = 0; it < 4; ++it){
      int idx = (tid + it*256) * 4;
      int r = idx >> 5, c = idx & 31;
      As[r][c+0] = (__bf16)fa[it].x; As[r][c+1] = (__bf16)fa[it].y;
      As[r][c+2] = (__bf16)fa[it].z; As[r][c+3] = (__bf16)fa[it].w;
    }
    #pragma unroll
    for (int i = 0; i < 4; ++i){
      union { unsigned long long q; __bf16 h[4]; } pk;
      #pragma unroll
      for (int j = 0; j < 4; ++j) pk.h[j] = (__bf16)fbv[j*4+i];
      *(unsigned long long*)(&Bt[cB + i][rB]) = pk.q;
    }
    __syncthreads();

    const bool more = (k0 + BK < K);
    if (more)   // overlap next tile's global loads with this tile's WMMAs
      stage_load(Ag, Bg, K, N, blockM, blockN, k0 + BK, tid, fa2, fbv2);

    v16bf afr[2], bfr[4];
    #pragma unroll
    for (int t = 0; t < 2; ++t){
      const __bf16* p0 = &As[waveM*32 + t*16 + l16][hi*8];
      #pragma unroll
      for (int i = 0; i < 8; ++i){ afr[t][i] = p0[i]; afr[t][8+i] = p0[16+i]; }
    }
    #pragma unroll
    for (int t = 0; t < 4; ++t){
      const __bf16* pb = &Bt[waveN*64 + t*16 + l16][hi*8];
      #pragma unroll
      for (int i = 0; i < 8; ++i){ bfr[t][i] = pb[i]; bfr[t][8+i] = pb[16+i]; }
    }
    #pragma unroll
    for (int i = 0; i < 2; ++i)
      #pragma unroll
      for (int j = 0; j < 4; ++j)
        acc[i][j] = __builtin_amdgcn_wmma_f32_16x16x32_bf16(
            false, afr[i], false, bfr[j], (short)0, acc[i][j], false, false);
    __syncthreads();

    if (more){
      #pragma unroll
      for (int it = 0; it < 4; ++it) fa[it] = fa2[it];
      #pragma unroll
      for (int it = 0; it < 16; ++it) fbv[it] = fbv2[it];
    }
  }

  #pragma unroll
  for (int i = 0; i < 2; ++i){
    #pragma unroll
    for (int j = 0; j < 4; ++j){
      int n = blockN + waveN*64 + j*16 + l16;
      float bv = bias[n];
      #pragma unroll
      for (int r = 0; r < 8; ++r){
        int m = blockM + waveM*32 + i*16 + hi*8 + r;
        float vv = acc[i][j][r] + bv;
        if (do_gelu) vv = gelu_tanh(vv);
        Dg[(size_t)m*N + n] = vv;
      }
    }
  }
}

// ---------------------------------------------------------------------------
// per-head RMSNorm + RoPE + scale -> bf16 head-major [H][S_TOT][HD]
__global__ void __launch_bounds__(256)
qk_prep(const float* __restrict__ simg, const float* __restrict__ sctx,
        const float* __restrict__ gimg, const float* __restrict__ gctx,
        const float* __restrict__ cosb, const float* __restrict__ sinb,
        void* outp, float qscale){
  __bf16* out = (__bf16*)outp;
  int gw = (blockIdx.x*256 + threadIdx.x) >> 5;
  int lane = threadIdx.x & 31;
  int h = gw % HEADS, s = gw / HEADS;
  const float* src; const float* g;
  if (s < S_TXT){ src = sctx + (size_t)s*DMODEL + h*HD; g = gctx; }
  else          { src = simg + (size_t)(s - S_TXT)*DMODEL + h*HD; g = gimg; }
  int d0 = lane*4;
  float v0 = src[d0+0], v1 = src[d0+1], v2 = src[d0+2], v3 = src[d0+3];
  float ss = v0*v0 + v1*v1 + v2*v2 + v3*v3;
  #pragma unroll
  for (int m = 16; m >= 1; m >>= 1) ss += __shfl_xor(ss, m, 32);
  float rr = rsqrtf(ss * (1.f/HD) + EPSV);
  float c0 = cosb[(size_t)s*HD + d0],   s0 = sinb[(size_t)s*HD + d0];
  float c1 = cosb[(size_t)s*HD + d0+2], s1 = sinb[(size_t)s*HD + d0+2];
  float t0 = v0*rr*g[d0], t1 = v1*rr*g[d0+1], t2 = v2*rr*g[d0+2], t3 = v3*rr*g[d0+3];
  __bf16* o = out + ((size_t)h*S_TOT + s)*HD + d0;
  o[0] = (__bf16)((t0*c0 - t1*s0) * qscale);
  o[1] = (__bf16)((t1*c0 + t0*s0) * qscale);
  o[2] = (__bf16)((t2*c1 - t3*s1) * qscale);
  o[3] = (__bf16)((t3*c1 + t2*s1) * qscale);
}

__global__ void __launch_bounds__(256)
v_prep(const float* __restrict__ simg, const float* __restrict__ sctx, void* outp){
  __bf16* out = (__bf16*)outp;
  int gw = (blockIdx.x*256 + threadIdx.x) >> 5;
  int lane = threadIdx.x & 31;
  int h = gw % HEADS, s = gw / HEADS;
  const float* src = (s < S_TXT) ? (sctx + (size_t)s*DMODEL + h*HD)
                                 : (simg + (size_t)(s - S_TXT)*DMODEL + h*HD);
  __bf16* o = out + ((size_t)h*S_TOT + s)*HD + lane*4;
  #pragma unroll
  for (int i = 0; i < 4; ++i) o[i] = (__bf16)src[lane*4 + i];
}

// ---------------------------------------------------------------------------
// flash attention, one (head, 64-query tile) per block, 4 waves
#define TQ 64
#define TK 64
#define KPITCH 136
#define VPITCH 72

__global__ void __launch_bounds__(128)
flash_attn(const void* qv, const void* kv, const void* vv, float* __restrict__ out){
  const __bf16* qj = (const __bf16*)qv;
  const __bf16* kj = (const __bf16*)kv;
  const __bf16* vj = (const __bf16*)vv;
  __shared__ __bf16 Qs[TQ][KPITCH];
  __shared__ __bf16 Ks[TK][KPITCH];
  __shared__ __bf16 Vt[HD][VPITCH];   // transposed: [d][key]
  __shared__ __bf16 Ps[TQ][VPITCH];
  const int tid = threadIdx.x, lane = tid & 31, w = tid >> 5;
  const int l16 = lane & 15, hi = lane >> 4;
  const int h = blockIdx.y, qt = blockIdx.x;
  const __bf16* qh = qj + (size_t)h*S_TOT*HD;
  const __bf16* kh = kj + (size_t)h*S_TOT*HD;
  const __bf16* vh = vj + (size_t)h*S_TOT*HD;

  // async raw-copy Q tile into LDS (ASYNCcnt path, no VGPR round-trip)
  #pragma unroll
  for (int it = 0; it < 8; ++it){
    int idx = tid + it*128;
    int row = idx >> 4, c8 = (idx & 15)*8;
    unsigned lo = (unsigned)(uintptr_t)(&Qs[row][c8]);
    unsigned go = (unsigned)(((qt*TQ + row)*HD + c8) * 2);
    async_copy_b128(lo, qh, go);
  }

  v8f oacc[8];
  #pragma unroll
  for (int t = 0; t < 8; ++t)
    #pragma unroll
    for (int e = 0; e < 8; ++e) oacc[t][e] = 0.f;
  float mrow[8], lrow[8];
  #pragma unroll
  for (int r = 0; r < 8; ++r){ mrow[r] = -3.0e38f; lrow[r] = 0.f; }

  for (int kt = 0; kt < S_TOT/TK; ++kt){
    __syncthreads();   // all waves done reading Ks/Vt from previous tile

    // K tile: single TDM DMA (wave 0), pads LDS rows to KPITCH itself
    if (w == 0){
      tdm_load_tile_64x128_bf16(kh + (size_t)(kt*TK)*HD,
                                (unsigned)(uintptr_t)(&Ks[0][0]));
      __builtin_amdgcn_s_wait_tensorcnt(0);
    }
    // V tile: 4 keys x 8 d per thread, packed b64 transposed stores
    #pragma unroll
    for (int j2 = 0; j2 < 2; ++j2){
      int q = tid + j2*128;
      int key4 = (q >> 4)*4;
      int d8   = (q & 15)*8;
      U4 u[4];
      #pragma unroll
      for (int kk = 0; kk < 4; ++kk)
        u[kk].u = *(const uint4*)(vh + (size_t)(kt*TK + key4 + kk)*HD + d8);
      #pragma unroll
      for (int i = 0; i < 8; ++i){
        union { unsigned long long qq; __bf16 h[4]; } pk;
        #pragma unroll
        for (int kk = 0; kk < 4; ++kk) pk.h[kk] = u[kk].h[i];
        *(unsigned long long*)(&Vt[d8 + i][key4]) = pk.qq;
      }
    }
    wait_async0();     // retires Q-tile async copies (first iteration)
    __syncthreads();

    // S = Q @ K^T  (16 q-rows per wave x 64 keys)
    v8f sacc[4];
    #pragma unroll
    for (int t = 0; t < 4; ++t)
      #pragma unroll
      for (int e = 0; e < 8; ++e) sacc[t][e] = 0.f;
    #pragma unroll
    for (int d0 = 0; d0 < HD; d0 += 32){
      v16bf a;
      { const __bf16* p0 = &Qs[w*16 + l16][d0 + hi*8];
        #pragma unroll
        for (int i = 0; i < 8; ++i){ a[i] = p0[i]; a[8+i] = p0[16+i]; } }
      #pragma unroll
      for (int t = 0; t < 4; ++t){
        v16bf b; const __bf16* pb = &Ks[t*16 + l16][d0 + hi*8];
        #pragma unroll
        for (int i = 0; i < 8; ++i){ b[i] = pb[i]; b[8+i] = pb[16+i]; }
        sacc[t] = __builtin_amdgcn_wmma_f32_16x16x32_bf16(
            false, a, false, b, (short)0, sacc[t], false, false);
      }
    }

    // online softmax over this key tile (C/D layout: row = hi*8 + r)
    #pragma unroll
    for (int r = 0; r < 8; ++r){
      float mx = fmaxf(fmaxf(sacc[0][r], sacc[1][r]), fmaxf(sacc[2][r], sacc[3][r]));
      #pragma unroll
      for (int msk = 8; msk >= 1; msk >>= 1) mx = fmaxf(mx, __shfl_xor(mx, msk, 32));
      float mn = fmaxf(mrow[r], mx);
      float corr = __expf(mrow[r] - mn);
      lrow[r] *= corr;
      #pragma unroll
      for (int t = 0; t < 8; ++t) oacc[t][r] *= corr;
      float rs = 0.f;
      #pragma unroll
      for (int t = 0; t < 4; ++t){
        float p = __expf(sacc[t][r] - mn);
        sacc[t][r] = p; rs += p;
      }
      #pragma unroll
      for (int msk = 8; msk >= 1; msk >>= 1) rs += __shfl_xor(rs, msk, 32);
      lrow[r] += rs; mrow[r] = mn;
    }

    // P -> LDS (wave-private rows, same-wave DS ordering suffices)
    #pragma unroll
    for (int t = 0; t < 4; ++t)
      #pragma unroll
      for (int r = 0; r < 8; ++r)
        Ps[w*16 + hi*8 + r][t*16 + l16] = (__bf16)sacc[t][r];

    // O += P @ V
    #pragma unroll
    for (int kc = 0; kc < TK; kc += 32){
      v16bf a;
      { const __bf16* p0 = &Ps[w*16 + l16][kc + hi*8];
        #pragma unroll
        for (int i = 0; i < 8; ++i){ a[i] = p0[i]; a[8+i] = p0[16+i]; } }
      #pragma unroll
      for (int t = 0; t < 8; ++t){
        v16bf b; const __bf16* pb = &Vt[t*16 + l16][kc + hi*8];
        #pragma unroll
        for (int i = 0; i < 8; ++i){ b[i] = pb[i]; b[8+i] = pb[16+i]; }
        oacc[t] = __builtin_amdgcn_wmma_f32_16x16x32_bf16(
            false, a, false, b, (short)0, oacc[t], false, false);
      }
    }
  }

  #pragma unroll
  for (int t = 0; t < 8; ++t)
    #pragma unroll
    for (int r = 0; r < 8; ++r){
      int s = qt*TQ + w*16 + hi*8 + r;
      out[(size_t)s*DMODEL + h*HD + t*16 + l16] = oacc[t][r] / lrow[r];
    }
}

// ---------------------------------------------------------------------------
extern "C" void kernel_launch(void* const* d_in, const int* in_sizes, int n_in,
                              void* d_out, int out_size, void* d_ws, size_t ws_size,
                              hipStream_t stream){
  (void)in_sizes; (void)n_in; (void)out_size; (void)ws_size;
  const float* x      = (const float*)d_in[0];
  const float* ctx    = (const float*)d_in[1];
  const float* temb   = (const float*)d_in[2];
  const float* rcos   = (const float*)d_in[3];
  const float* rsin   = (const float*)d_in[4];
  const float* ada_xw = (const float*)d_in[5];
  const float* ada_xb = (const float*)d_in[6];
  const float* ada_cw = (const float*)d_in[7];
  const float* ada_cb = (const float*)d_in[8];
  const float* wq  = (const float*)d_in[9];  const float* bq  = (const float*)d_in[10];
  const float* wk  = (const float*)d_in[11]; const float* bk  = (const float*)d_in[12];
  const float* wv  = (const float*)d_in[13]; const float* bv  = (const float*)d_in[14];
  const float* wo  = (const float*)d_in[15]; const float* bo  = (const float*)d_in[16];
  const float* wqc = (const float*)d_in[17]; const float* bqc = (const float*)d_in[18];
  const float* wkc = (const float*)d_in[19]; const float* bkc = (const float*)d_in[20];
  const float* wvc = (const float*)d_in[21]; const float* bvc = (const float*)d_in[22];
  const float* wadd= (const float*)d_in[23]; const float* badd= (const float*)d_in[24];
  const float* gq  = (const float*)d_in[25]; const float* gk  = (const float*)d_in[26];
  const float* gqc = (const float*)d_in[27]; const float* gkc = (const float*)d_in[28];
  const float* ff1w = (const float*)d_in[29]; const float* ff1b = (const float*)d_in[30];
  const float* ff2w = (const float*)d_in[31]; const float* ff2b = (const float*)d_in[32];
  const float* fc1w = (const float*)d_in[33]; const float* fc1b = (const float*)d_in[34];
  const float* fc2w = (const float*)d_in[35]; const float* fc2b = (const float*)d_in[36];
  float* outp = (float*)d_out;
  float* W = (float*)d_ws;

  const size_t SZX = (size_t)S_IMG*DMODEL;
  const size_t SZC = (size_t)S_TXT*DMODEL;
  const size_t SZHSD_F = (size_t)HEADS*S_TOT*HD/2; // bf16 buffer in float units

  size_t off = 0;
  auto alloc = [&](size_t nf){ size_t r = off; off += (nf + 63) & ~(size_t)63; return r; };
  size_t o_z    = alloc(DMODEL);
  size_t o_modx = alloc(6*DMODEL);
  size_t o_modc = alloc(6*DMODEL);
  size_t o_nx   = alloc(SZX);
  size_t o_nc   = alloc(SZC);
  size_t o_q  = alloc(SZX), o_k  = alloc(SZX), o_v  = alloc(SZX);
  size_t o_qc = alloc(SZC), o_kc = alloc(SZC), o_vc = alloc(SZC);
  size_t o_qj = alloc(SZHSD_F), o_kj = alloc(SZHSD_F), o_vj = alloc(SZHSD_F);
  size_t o_ao = alloc((size_t)S_TOT*DMODEL);
  size_t o_ia = alloc(SZX), o_ca = alloc(SZC);
  size_t o_hid = alloc(SZX), o_nf = alloc(SZX);
  size_t o_g1  = alloc((size_t)S_IMG*4*DMODEL), o_f2 = alloc(SZX);
  size_t o_enc = alloc(SZC), o_ncf = alloc(SZC);
  size_t o_gc1 = alloc((size_t)S_TXT*4*DMODEL), o_fc2 = alloc(SZC);

  // 1) z = silu(temb)
  silu_kernel<<<DMODEL/256, 256, 0, stream>>>(temb, W+o_z, DMODEL);
  // 2) adaLN modulations
  ada_gemv<<<(6*DMODEL)/256, 256, 0, stream>>>(W+o_z, ada_xw, ada_xb, W+o_modx, DMODEL, 6*DMODEL);
  ada_gemv<<<(6*DMODEL)/256, 256, 0, stream>>>(W+o_z, ada_cw, ada_cb, W+o_modc, DMODEL, 6*DMODEL);
  // 3) pre-attention LN + modulate (shift=mod[0:D], scale=mod[D:2D])
  ln_mod<<<S_IMG, 256, 0, stream>>>(x,   W+o_modx+DMODEL, W+o_modx, W+o_nx);
  ln_mod<<<S_TXT, 256, 0, stream>>>(ctx, W+o_modc+DMODEL, W+o_modc, W+o_nc);
  // 4) QKV projections (bf16 WMMA GEMMs)
  dim3 gI(DMODEL/BN, S_IMG/BM), gC(DMODEL/BN, S_TXT/BM);
  gemm_bf16<<<gI, 256, 0, stream>>>(W+o_nx, wq,  bq,  W+o_q,  S_IMG, DMODEL, DMODEL, 0);
  gemm_bf16<<<gI, 256, 0, stream>>>(W+o_nx, wk,  bk,  W+o_k,  S_IMG, DMODEL, DMODEL, 0);
  gemm_bf16<<<gI, 256, 0, stream>>>(W+o_nx, wv,  bv,  W+o_v,  S_IMG, DMODEL, DMODEL, 0);
  gemm_bf16<<<gC, 256, 0, stream>>>(W+o_nc, wqc, bqc, W+o_qc, S_TXT, DMODEL, DMODEL, 0);
  gemm_bf16<<<gC, 256, 0, stream>>>(W+o_nc, wkc, bkc, W+o_kc, S_TXT, DMODEL, DMODEL, 0);
  gemm_bf16<<<gC, 256, 0, stream>>>(W+o_nc, wvc, bvc, W+o_vc, S_TXT, DMODEL, DMODEL, 0);
  // 5) RMS + RoPE prep into bf16 head-major; fold 1/sqrt(HD) into q
  int prep_blocks = (S_TOT*HEADS)/8;     // one wave per (s,h)
  qk_prep<<<prep_blocks, 256, 0, stream>>>(W+o_q, W+o_qc, gq, gqc, rcos, rsin,
                                           (void*)(W+o_qj), 0.08838834764831845f);
  qk_prep<<<prep_blocks, 256, 0, stream>>>(W+o_k, W+o_kc, gk, gkc, rcos, rsin,
                                           (void*)(W+o_kj), 1.0f);
  v_prep<<<prep_blocks, 256, 0, stream>>>(W+o_v, W+o_vc, (void*)(W+o_vj));
  // 6) joint flash attention
  dim3 gA(S_TOT/TQ, HEADS);
  flash_attn<<<gA, 128, 0, stream>>>((void*)(W+o_qj), (void*)(W+o_kj), (void*)(W+o_vj), W+o_ao);
  // 7) output projections (ctx rows first in joint order)
  gemm_bf16<<<gC, 256, 0, stream>>>(W+o_ao,                      wadd, badd, W+o_ca, S_TXT, DMODEL, DMODEL, 0);
  gemm_bf16<<<gI, 256, 0, stream>>>(W+o_ao+(size_t)S_TXT*DMODEL, wo,   bo,   W+o_ia, S_IMG, DMODEL, DMODEL, 0);
  // 8) residual with gate_msa (mod[2D:3D])
  long long totI = (long long)SZX, totC = (long long)SZC;
  res_gate<<<(unsigned)((totI+255)/256), 256, 0, stream>>>(x,   W+o_ia, W+o_modx+2*DMODEL, W+o_hid, totI, DMODEL);
  res_gate<<<(unsigned)((totC+255)/256), 256, 0, stream>>>(ctx, W+o_ca, W+o_modc+2*DMODEL, W+o_enc, totC, DMODEL);
  // 9) MLP LN + modulate (shift=mod[3D:4D], scale=mod[4D:5D])
  ln_mod<<<S_IMG, 256, 0, stream>>>(W+o_hid, W+o_modx+4*DMODEL, W+o_modx+3*DMODEL, W+o_nf);
  ln_mod<<<S_TXT, 256, 0, stream>>>(W+o_enc, W+o_modc+4*DMODEL, W+o_modc+3*DMODEL, W+o_ncf);
  // 10) MLPs with fused GELU
  dim3 gI1(4*DMODEL/BN, S_IMG/BM), gC1(4*DMODEL/BN, S_TXT/BM);
  gemm_bf16<<<gI1, 256, 0, stream>>>(W+o_nf,  ff1w, ff1b, W+o_g1,  S_IMG, 4*DMODEL, DMODEL, 1);
  gemm_bf16<<<gI,  256, 0, stream>>>(W+o_g1,  ff2w, ff2b, W+o_f2,  S_IMG, DMODEL, 4*DMODEL, 0);
  gemm_bf16<<<gC1, 256, 0, stream>>>(W+o_ncf, fc1w, fc1b, W+o_gc1, S_TXT, 4*DMODEL, DMODEL, 1);
  gemm_bf16<<<gC,  256, 0, stream>>>(W+o_gc1, fc2w, fc2b, W+o_fc2, S_TXT, DMODEL, 4*DMODEL, 0);
  // 11) final gated residuals -> d_out = [enc (256x3072), hid (2048x3072)]
  res_gate<<<(unsigned)((totC+255)/256), 256, 0, stream>>>(W+o_enc, W+o_fc2, W+o_modc+5*DMODEL, outp, totC, DMODEL);
  res_gate<<<(unsigned)((totI+255)/256), 256, 0, stream>>>(W+o_hid, W+o_f2,  W+o_modx+5*DMODEL, outp + SZC, totI, DMODEL);
}
